// HybridFCL_65481071396920
// MI455X (gfx1250) — matrix operand order repaired and checked by reading, not compiled, not run on hardware
//
#include <hip/hip_runtime.h>
#include <hip/hip_bf16.h>
#include <math.h>

// Problem constants (from reference setup_inputs)
#define B_  32
#define T_  512
#define D_  256
#define P_  512
#define H_  512
#define G_  (4 * H_)      // 2048
#define N_  (B_ * T_)     // 16384

// GEMM tiling
#define MTILE 128
#define NTILE 64
#define KTILE 32
#define KPAD  40          // 32 bf16 + 8 bf16 pad (80 B row: 16B-aligned, conflict-minimal)

typedef __bf16 bf16_t;
typedef __attribute__((ext_vector_type(16))) __bf16 bf16x16;
typedef __attribute__((ext_vector_type(8)))  float  floatx8;

union FragU { bf16x16 v; float4 q[2]; };

__device__ __forceinline__ bf16_t f2bf(float f) {
    union { float f; unsigned u; } v; v.f = f;
    unsigned r = (v.u + 0x7FFFu + ((v.u >> 16) & 1u)) >> 16;
    unsigned short s = (unsigned short)r;
    bf16_t out;
    __builtin_memcpy(&out, &s, 2);
    return out;
}

// ---------------------------------------------------------------------------
// Async global -> LDS copy (gfx1250): 16 bytes per lane, tracked by ASYNCcnt.
// VDST carries the LDS byte offset (low 32 bits of the flat LDS address).
// ---------------------------------------------------------------------------
__device__ __forceinline__ void async_b128(void* lds_dst, const void* gsrc) {
    unsigned loff = (unsigned)(unsigned long long)lds_dst;
    asm volatile("global_load_async_to_lds_b128 %0, %1, off"
                 :: "v"(loff), "v"(gsrc) : "memory");
}
__device__ __forceinline__ void wait_async0() {
    asm volatile("s_wait_asynccnt 0x0" ::: "memory");
}

// ---------------------------------------------------------------------------
// Fragment loaders
// A frag (16x32, MxK) ISA layout: lane L<16 -> M=L, K {0..7}U{16..23};
// lane L>=16 -> M=L-16, K {8..15}U{24..31}
// B frag (32x16, KxN) from row-major (N x K) weights: B(k,n)=src[n*ld+k]
// ---------------------------------------------------------------------------
__device__ __forceinline__ bf16x16 lds_frag_a(const bf16_t* sA, int mloc) {
    int lane = threadIdx.x & 31;
    const bf16_t* p = sA + (mloc + (lane & 15)) * KPAD + ((lane >> 4) << 3);
    FragU f;
    f.q[0] = *(const float4*)(p);
    f.q[1] = *(const float4*)(p + 16);
    return f.v;
}
__device__ __forceinline__ bf16x16 lds_frag_b(const bf16_t* sB, int nloc) {
    int lane = threadIdx.x & 31;
    const bf16_t* p = sB + (nloc + (lane & 15)) * KPAD + ((lane >> 4) << 3);
    FragU f;
    f.q[0] = *(const float4*)(p);
    f.q[1] = *(const float4*)(p + 16);
    return f.v;
}
// Direct-from-global variants (used by the small LSTM-step GEMM)
__device__ __forceinline__ bf16x16 g_frag_a(const bf16_t* __restrict__ src, int ld,
                                            int m0, int k0) {
    int lane = threadIdx.x & 31;
    const bf16_t* p = src + (size_t)(m0 + (lane & 15)) * ld + k0 + ((lane >> 4) << 3);
    FragU f;
    f.q[0] = *(const float4*)(p);
    f.q[1] = *(const float4*)(p + 16);
    return f.v;
}
__device__ __forceinline__ bf16x16 g_frag_b(const bf16_t* __restrict__ src, int ld,
                                            int n0, int k0) {
    int lane = threadIdx.x & 31;
    const bf16_t* p = src + (size_t)(n0 + (lane & 15)) * ld + k0 + ((lane >> 4) << 3);
    FragU f;
    f.q[0] = *(const float4*)(p);
    f.q[1] = *(const float4*)(p + 16);
    return f.v;
}

#define WMMA_BF16(A, Bm, C) \
    __builtin_amdgcn_wmma_f32_16x16x32_bf16(false, (A), false, (Bm), (short)0, (C), false, false)

// ---------------------------------------------------------------------------
// Staged GEMM core: block = 256 thr (8 waves as 4M x 2N), tile 128x64,
// double-buffered async LDS staging, 2x2 register blocking per wave.
// ---------------------------------------------------------------------------
__device__ __forceinline__ void stage_tile(const bf16_t* __restrict__ A, int lda, int m0,
                                           const bf16_t* __restrict__ W, int ldb, int n0,
                                           int k0, bf16_t* sA, bf16_t* sB) {
    int tid = threadIdx.x;
    // A tile: 128 rows x 4 segments (8 bf16 = 16B each) = 512 chunks, 2/thread
    int row = tid >> 2, seg = tid & 3;
    async_b128(sA + row * KPAD + seg * 8,
               A + (size_t)(m0 + row) * lda + k0 + seg * 8);
    row += 64;
    async_b128(sA + row * KPAD + seg * 8,
               A + (size_t)(m0 + row) * lda + k0 + seg * 8);
    // B tile: 64 rows x 4 segments = 256 chunks, 1/thread
    row = tid >> 2;
    async_b128(sB + row * KPAD + seg * 8,
               W + (size_t)(n0 + row) * ldb + k0 + seg * 8);
}

__device__ __forceinline__ void compute_step(const bf16_t* sA, const bf16_t* sB,
                                             int mw, int nw, floatx8 acc[2][2]) {
    bf16x16 a0 = lds_frag_a(sA, mw * 32);
    bf16x16 a1 = lds_frag_a(sA, mw * 32 + 16);
    bf16x16 b0 = lds_frag_b(sB, nw * 32);
    bf16x16 b1 = lds_frag_b(sB, nw * 32 + 16);
    acc[0][0] = WMMA_BF16(a0, b0, acc[0][0]);
    acc[0][1] = WMMA_BF16(a0, b1, acc[0][1]);
    acc[1][0] = WMMA_BF16(a1, b0, acc[1][0]);
    acc[1][1] = WMMA_BF16(a1, b1, acc[1][1]);
}

// Shared pipeline body: computes acc[2][2] for this wave's 32x32 sub-tile.
#define GEMM_PIPELINE(Aptr, lda, Wptr, ldb, KSTEPS)                                   \
    __shared__ bf16_t sA[2][MTILE * KPAD];                                            \
    __shared__ bf16_t sB[2][NTILE * KPAD];                                            \
    int w    = threadIdx.x >> 5;                                                      \
    int lane = threadIdx.x & 31;                                                      \
    int mw = w >> 1, nw = w & 1;                                                      \
    int m0 = blockIdx.x * MTILE;                                                      \
    int n0 = blockIdx.y * NTILE;                                                      \
    floatx8 acc[2][2];                                                                \
    acc[0][0] = (floatx8){}; acc[0][1] = (floatx8){};                                 \
    acc[1][0] = (floatx8){}; acc[1][1] = (floatx8){};                                 \
    stage_tile(Aptr, lda, m0, Wptr, ldb, n0, 0, sA[0], sB[0]);                        \
    wait_async0();                                                                    \
    __syncthreads();                                                                  \
    int buf = 0;                                                                      \
    for (int kt = 0; kt < (KSTEPS); ++kt) {                                           \
        if (kt + 1 < (KSTEPS))                                                        \
            stage_tile(Aptr, lda, m0, Wptr, ldb, n0, (kt + 1) * KTILE,                \
                       sA[buf ^ 1], sB[buf ^ 1]);                                     \
        compute_step(sA[buf], sB[buf], mw, nw, acc);                                  \
        wait_async0();                                                                \
        __syncthreads();                                                              \
        buf ^= 1;                                                                     \
    }                                                                                 \
    (void)lane;

// ---------------------------------------------------------------------------
// Prep kernels
// ---------------------------------------------------------------------------
__global__ void k_cvt_bf16(const float* __restrict__ in, bf16_t* __restrict__ out, int n) {
    int i = blockIdx.x * blockDim.x + threadIdx.x;
    if (i < n) out[i] = f2bf(in[i]);
}

__global__ void k_rownorm256(const float* __restrict__ in, float* __restrict__ out) {
    __shared__ float s[256];
    const float* p = in + (size_t)blockIdx.x * 256;
    float x = p[threadIdx.x];
    s[threadIdx.x] = x * x;
    __syncthreads();
    for (int off = 128; off > 0; off >>= 1) {
        if (threadIdx.x < off) s[threadIdx.x] += s[threadIdx.x + off];
        __syncthreads();
    }
    if (threadIdx.x == 0) out[blockIdx.x] = s[0];
}

__global__ void k_init_state(const float* __restrict__ bi, const float* __restrict__ bh,
                             float* __restrict__ b4, bf16_t* __restrict__ h0,
                             float* __restrict__ c) {
    int i = blockIdx.x * blockDim.x + threadIdx.x;
    if (i < G_) b4[i] = bi[i] + bh[i];
    if (i < B_ * H_) { c[i] = 0.0f; h0[i] = f2bf(0.0f); }
}

// ---------------------------------------------------------------------------
// GEMM 1: RBF  feats = exp(-max(||x||^2 + ||p||^2 - 2 x.p, 0))   (N x P, K=D)
// ---------------------------------------------------------------------------
__global__ void k_rbf_gemm(const bf16_t* __restrict__ xb, const bf16_t* __restrict__ pb,
                           const float* __restrict__ x2, const float* __restrict__ p2,
                           bf16_t* __restrict__ feats) {
    GEMM_PIPELINE(xb, D_, pb, D_, D_ / KTILE)
#pragma unroll
    for (int ti = 0; ti < 2; ++ti)
#pragma unroll
        for (int tj = 0; tj < 2; ++tj) {
            int mrow = m0 + mw * 32 + ti * 16 + ((lane >> 4) << 3);
            int n    = n0 + nw * 32 + tj * 16 + (lane & 15);
            float pn = p2[n];
#pragma unroll
            for (int r = 0; r < 8; ++r) {
                int m = mrow + r;
                float sq = x2[m] + pn - 2.0f * acc[ti][tj][r];
                sq = fmaxf(sq, 0.0f);
                feats[(size_t)m * P_ + n] = f2bf(__expf(-sq));
            }
        }
}

// ---------------------------------------------------------------------------
// GEMM 2: feats2 = feats @ W_lin^T + b_lin   (N x H, K=P)
// ---------------------------------------------------------------------------
__global__ void k_lin_gemm(const bf16_t* __restrict__ feats, const bf16_t* __restrict__ wl,
                           const float* __restrict__ bl, bf16_t* __restrict__ feats2) {
    GEMM_PIPELINE(feats, P_, wl, P_, P_ / KTILE)
#pragma unroll
    for (int ti = 0; ti < 2; ++ti)
#pragma unroll
        for (int tj = 0; tj < 2; ++tj) {
            int mrow = m0 + mw * 32 + ti * 16 + ((lane >> 4) << 3);
            int n    = n0 + nw * 32 + tj * 16 + (lane & 15);
            float bn = bl[n];
#pragma unroll
            for (int r = 0; r < 8; ++r) {
                int m = mrow + r;
                feats2[(size_t)m * H_ + n] = f2bf(acc[ti][tj][r] + bn);
            }
        }
}

// ---------------------------------------------------------------------------
// GEMM 3: xs[t][b][g] = feats2[b*T+t] @ W_ih^T + (b_ih+b_hh)   (N x 4H, K=H)
// ---------------------------------------------------------------------------
__global__ void k_ih_gemm(const bf16_t* __restrict__ feats2, const bf16_t* __restrict__ wih,
                          const float* __restrict__ b4, float* __restrict__ xs) {
    GEMM_PIPELINE(feats2, H_, wih, H_, H_ / KTILE)
#pragma unroll
    for (int ti = 0; ti < 2; ++ti)
#pragma unroll
        for (int tj = 0; tj < 2; ++tj) {
            int mrow = m0 + mw * 32 + ti * 16 + ((lane >> 4) << 3);
            int n    = n0 + nw * 32 + tj * 16 + (lane & 15);
            float bn = b4[n];
#pragma unroll
            for (int r = 0; r < 8; ++r) {
                int m = mrow + r;
                int bidx = m >> 9;          // m / T_
                int t    = m & (T_ - 1);    // m % T_
                xs[((size_t)t * B_ + bidx) * G_ + n] = acc[ti][tj][r] + bn;
            }
        }
}

// ---------------------------------------------------------------------------
// LSTM step t: gates = xs[t] + h_in @ W_hh^T  (32 x 2048, K=512), fused
// pointwise. Block owns 16 hidden units j and ALL 4 gates for them:
// 8 waves = 2(M=batch) x 4(gate); gates meet in LDS for the cell update.
// ---------------------------------------------------------------------------
__global__ void k_lstm_step(const float* __restrict__ xs, const bf16_t* __restrict__ whh,
                            const bf16_t* __restrict__ h_in, bf16_t* __restrict__ h_out,
                            float* __restrict__ c, float* __restrict__ out, int t) {
    __shared__ float gl[4][B_][16];
    int w    = threadIdx.x >> 5;
    int lane = threadIdx.x & 31;
    int q  = w >> 1;                 // gate: 0=i 1=f 2=g 3=o
    int m0 = (w & 1) * 16;           // batch tile
    int j0 = blockIdx.x * 16;        // hidden-unit slice
    int nC = q * H_ + j0;            // column base in 4H gate space
    floatx8 acc = {};
#pragma unroll
    for (int k0 = 0; k0 < H_; k0 += KTILE) {
        bf16x16 a = g_frag_a(h_in, H_, m0, k0);
        bf16x16 b = g_frag_b(whh, H_, nC, k0);
        acc = WMMA_BF16(a, b, acc);
    }
    const float* xrow = xs + (size_t)t * B_ * G_;
    int jl   = lane & 15;
    int mrow = m0 + ((lane >> 4) << 3);
#pragma unroll
    for (int r = 0; r < 8; ++r) {
        int m = mrow + r;
        gl[q][m][jl] = acc[r] + xrow[(size_t)m * G_ + nC + jl];
    }
    __syncthreads();
    for (int idx = threadIdx.x; idx < B_ * 16; idx += blockDim.x) {
        int m  = idx >> 4;
        int jj = idx & 15;
        int j  = j0 + jj;
        float gi = gl[0][m][jj];
        float gf = gl[1][m][jj];
        float gg = gl[2][m][jj];
        float go = gl[3][m][jj];
        gi = 1.0f / (1.0f + __expf(-gi));
        gf = 1.0f / (1.0f + __expf(-gf));
        go = 1.0f / (1.0f + __expf(-go));
        gg = tanhf(gg);
        float cn = gf * c[m * H_ + j] + gi * gg;
        c[m * H_ + j] = cn;
        float h = go * tanhf(cn);
        out[(size_t)m * T_ * H_ + (size_t)t * H_ + j] = h;   // (B,T,H)
        h_out[m * H_ + j] = f2bf(h);
    }
}

// ---------------------------------------------------------------------------
extern "C" void kernel_launch(void* const* d_in, const int* in_sizes, int n_in,
                              void* d_out, int out_size, void* d_ws, size_t ws_size,
                              hipStream_t stream) {
    const float* x      = (const float*)d_in[0];
    const float* protos = (const float*)d_in[1];
    const float* W_lin  = (const float*)d_in[2];
    const float* b_lin  = (const float*)d_in[3];
    const float* W_ih   = (const float*)d_in[4];
    const float* W_hh   = (const float*)d_in[5];
    const float* b_ih   = (const float*)d_in[6];
    const float* b_hh   = (const float*)d_in[7];
    float* out = (float*)d_out;

    char* p = (char*)d_ws;
    auto alloc = [&](size_t bytes) -> char* {
        char* r = p;
        p += (bytes + 255) & ~(size_t)255;
        return r;
    };
    bf16_t* xb     = (bf16_t*)alloc((size_t)N_ * D_ * 2);
    bf16_t* pb     = (bf16_t*)alloc((size_t)P_ * D_ * 2);
    bf16_t* wlinb  = (bf16_t*)alloc((size_t)H_ * P_ * 2);
    bf16_t* wihb   = (bf16_t*)alloc((size_t)G_ * H_ * 2);
    bf16_t* whhb   = (bf16_t*)alloc((size_t)G_ * H_ * 2);
    float*  x2     = (float*)alloc((size_t)N_ * 4);
    float*  p2     = (float*)alloc((size_t)P_ * 4);
    float*  b4     = (float*)alloc((size_t)G_ * 4);
    bf16_t* feats  = (bf16_t*)alloc((size_t)N_ * P_ * 2);
    bf16_t* feats2 = (bf16_t*)alloc((size_t)N_ * H_ * 2);
    float*  xs     = (float*)alloc((size_t)T_ * B_ * G_ * 4);
    bf16_t* hb0    = (bf16_t*)alloc((size_t)B_ * H_ * 2);
    bf16_t* hb1    = (bf16_t*)alloc((size_t)B_ * H_ * 2);
    float*  c      = (float*)alloc((size_t)B_ * H_ * 4);

    const int thr = 256;
    k_cvt_bf16<<<(N_ * D_ + thr - 1) / thr, thr, 0, stream>>>(x, xb, N_ * D_);
    k_cvt_bf16<<<(P_ * D_ + thr - 1) / thr, thr, 0, stream>>>(protos, pb, P_ * D_);
    k_cvt_bf16<<<(H_ * P_ + thr - 1) / thr, thr, 0, stream>>>(W_lin, wlinb, H_ * P_);
    k_cvt_bf16<<<(G_ * H_ + thr - 1) / thr, thr, 0, stream>>>(W_ih, wihb, G_ * H_);
    k_cvt_bf16<<<(G_ * H_ + thr - 1) / thr, thr, 0, stream>>>(W_hh, whhb, G_ * H_);
    k_rownorm256<<<N_, 256, 0, stream>>>(x, x2);
    k_rownorm256<<<P_, 256, 0, stream>>>(protos, p2);
    k_init_state<<<(B_ * H_ + thr - 1) / thr, thr, 0, stream>>>(b_ih, b_hh, b4, hb0, c);

    k_rbf_gemm<<<dim3(N_ / MTILE, P_ / NTILE), 256, 0, stream>>>(xb, pb, x2, p2, feats);
    k_lin_gemm<<<dim3(N_ / MTILE, H_ / NTILE), 256, 0, stream>>>(feats, wlinb, b_lin, feats2);
    k_ih_gemm<<<dim3(N_ / MTILE, G_ / NTILE), 256, 0, stream>>>(feats2, wihb, b4, xs);

    for (int t = 0; t < T_; ++t) {
        bf16_t* hin  = (t & 1) ? hb1 : hb0;
        bf16_t* hout = (t & 1) ? hb0 : hb1;
        k_lstm_step<<<H_ / 16, 256, 0, stream>>>(xs, whhb, hin, hout, c, out, t);
    }
}
